// MultiHeadSelfAttention_81492709474649
// MI455X (gfx1250) — matrix operand order repaired and checked by reading, not compiled, not run on hardware
//
#include <hip/hip_runtime.h>

typedef _Float16 h16;
typedef __attribute__((ext_vector_type(16))) _Float16 v16h;
typedef __attribute__((ext_vector_type(8)))  _Float16 v8h;
typedef __attribute__((ext_vector_type(4)))  _Float16 v4h;
typedef __attribute__((ext_vector_type(8)))  float    v8f;

constexpr int kB = 2;
constexpr int kS = 2048;
constexpr int kE = 1024;     // embed dim (= K of projection GEMMs)
constexpr int kN = 1024;     // hidden dim (= N of projection GEMMs)
constexpr int kH = 16;
constexpr int kD = 64;
constexpr int kM = kB * kS;  // 4096 rows

#if __has_builtin(__builtin_amdgcn_exp2f)
#define EXP2F __builtin_amdgcn_exp2f
#else
#define EXP2F exp2f
#endif
#if __has_builtin(__builtin_amdgcn_rcpf)
#define RCPF __builtin_amdgcn_rcpf
#else
#define RCPF(x) (1.0f / (x))
#endif

__device__ __forceinline__ v8f wmma_f16(v16h a, v16h b, v8f c) {
  return __builtin_amdgcn_wmma_f32_16x16x32_f16(false, a, false, b,
                                                (short)0, c, false, false);
}

// 16x32 f16 A-tile from row-major source (stride ld halfs).  Contiguous k order used here (same as load_b32x16):
// lane = r + 16*hf holds row r; elements 0..15 = K hf*16..hf*16+15
__device__ __forceinline__ v16h load_a16x32(const h16* __restrict__ p, int row0,
                                            int k0, int ld, int r, int hf) {
  const h16* base = p + (size_t)(row0 + r) * ld + k0 + hf * 16;
  v8h c0 = *(const v8h*)(base);
  v8h c1 = *(const v8h*)(base + 8);
  v16h a;
#pragma unroll
  for (int e = 0; e < 8; ++e) { a[e] = c0[e]; a[8 + e] = c1[e]; }
  return a;
}

// 32x16 B-tile from the TRANSPOSED operand Bt (row-major [N][K], stride ld):
// B(k,n) = Bt[n0+n][k0+k].  Lane = n + 16*(k/16); elements = K hf*16..+15.
__device__ __forceinline__ v16h load_b32x16(const h16* __restrict__ bt, int n0,
                                            int k0, int ld, int r, int hf) {
  return *(const v16h*)(bt + (size_t)(n0 + r) * ld + k0 + hf * 16);
}

// ---------------------------------------------------------------------------
// Stage 0a: f32 -> f16 convert (X), 4 elements/thread
__global__ void cvt_f32_to_f16(const float* __restrict__ in,
                               h16* __restrict__ out, int n4) {
  int i = (blockIdx.x * blockDim.x + threadIdx.x);
  if (i < n4) {
    float4 v = *(const float4*)(in + i * 4);
    v4h o; o[0] = (h16)v.x; o[1] = (h16)v.y; o[2] = (h16)v.z; o[3] = (h16)v.w;
    *(v4h*)(out + i * 4) = o;
  }
}

// Stage 0b: tiled transpose + convert:  out[n][k] = (f16) in[k][n], 1024x1024
__global__ void transpose_to_f16(const float* __restrict__ in,
                                 h16* __restrict__ out) {
  __shared__ float tile[32][33];
  int bx = blockIdx.x * 32, by = blockIdx.y * 32;
  int tx = threadIdx.x, ty = threadIdx.y;            // block 32x8
#pragma unroll
  for (int i = ty; i < 32; i += 8)
    tile[i][tx] = in[(size_t)(by + i) * kE + bx + tx];
  __syncthreads();
#pragma unroll
  for (int i = ty; i < 32; i += 8)
    out[(size_t)(bx + i) * kE + by + tx] = (h16)tile[tx][i];
}

// ---------------------------------------------------------------------------
// Stage 1: projection GEMM  C[m][n] = Xh[m][:] . W[:, n] + bias[n]
// Wt is W transposed ([N][K] f16).  Output written head-major:
//   Out [bh][s][d]   (bh = b*16+h, n = h*64+d)
//   OutT[bh][d][s]   (WRITE_T only; used for V)
// Block: 128 thr = 4 waves (2x2), wave tile 32x64, WG tile 64x128.
// Inner loop: 8 WMMA per 12 x b128 loads (A reused across 4 B tiles).
template <bool WRITE_T>
__global__ __launch_bounds__(128) void proj_gemm(
    const h16* __restrict__ Xh, const h16* __restrict__ Wt,
    const float* __restrict__ bias, h16* __restrict__ Out,
    h16* __restrict__ OutT) {
  const int lane = threadIdx.x & 31;
  const int wave = threadIdx.x >> 5;
  const int r = lane & 15, hf = lane >> 4;
  const int m0 = blockIdx.x * 64 + (wave >> 1) * 32;
  const int n0 = blockIdx.y * 128 + (wave & 1) * 64;

  v8f acc[2][4] = {};
  for (int k0 = 0; k0 < kE; k0 += 32) {
    v16h a[2], b[4];
#pragma unroll
    for (int i = 0; i < 2; ++i) a[i] = load_a16x32(Xh, m0 + i * 16, k0, kE, r, hf);
#pragma unroll
    for (int j = 0; j < 4; ++j) b[j] = load_b32x16(Wt, n0 + j * 16, k0, kE, r, hf);
#pragma unroll
    for (int i = 0; i < 2; ++i)
#pragma unroll
      for (int j = 0; j < 4; ++j) acc[i][j] = wmma_f16(a[i], b[j], acc[i][j]);
  }
#pragma unroll
  for (int j = 0; j < 4; ++j) {
    const int ncol = n0 + j * 16 + r;       // C/D: lane&15 = column
    const int hh = ncol >> 6, d = ncol & 63;
    const float bv = bias[ncol];
#pragma unroll
    for (int t = 0; t < 2; ++t)
#pragma unroll
      for (int reg = 0; reg < 8; ++reg) {
        const int m = m0 + t * 16 + 8 * hf + reg;  // C/D: row = reg + 8*half
        const int bi = m >> 11, s = m & 2047;
        const h16 val = (h16)(acc[t][j][reg] + bv);
        const int bh = bi * kH + hh;
        Out[((size_t)bh * kS + s) * kD + d] = val;
        if constexpr (WRITE_T)
          OutT[((size_t)bh * kD + d) * kS + s] = val;
      }
  }
}

// ---------------------------------------------------------------------------
// Stage 2: flash attention, scores = Q.V^T (source bug), softmax, O = P.V
// Grid: (B*H, S/64); block 128 thr = 4 waves, each wave owns 16 queries.
// All 8 B-tile loads of a key block are hoisted to the iteration top so the
// softmax VALU block hides their latency before the P.V WMMAs consume them.
__global__ __launch_bounds__(128) void attn_kernel(
    const h16* __restrict__ Q, const h16* __restrict__ V,
    const h16* __restrict__ Vt, float* __restrict__ Out) {
  __shared__ h16 plds[4][16 * 32];
  const int lane = threadIdx.x & 31;
  const int wave = threadIdx.x >> 5;
  const int r = lane & 15, hf = lane >> 4;

  const int bh = blockIdx.x;
  const int q0 = blockIdx.y * 64 + wave * 16;

  const h16* Qp  = Q  + (size_t)bh * kS * kD;
  const h16* Vp  = V  + (size_t)bh * kS * kD;
  const h16* Vtp = Vt + (size_t)bh * kD * kS;
  h16* P = plds[wave];

  v16h aq[2];
#pragma unroll
  for (int c = 0; c < 2; ++c) aq[c] = load_a16x32(Qp, q0, c * 32, kD, r, hf);

  v8f o[4] = {};
  float mrow[8], lrow[8];
#pragma unroll
  for (int t = 0; t < 8; ++t) { mrow[t] = -1e30f; lrow[t] = 0.f; }
  const float sc = 0.125f * 1.44269504088896f;  // 1/sqrt(64) * log2(e)

  for (int key0 = 0; key0 < kS; key0 += 32) {
    // ---- all loads for this key block, issued up front -------------------
    v16h bs[2][2];  // score B: [key tile][d chunk]
#pragma unroll
    for (int c = 0; c < 2; ++c) {
      bs[0][c] = load_b32x16(Vp, key0,      c * 32, kD, r, hf);
      bs[1][c] = load_b32x16(Vp, key0 + 16, c * 32, kD, r, hf);
    }
    v16h bv[4];     // P.V B tiles (independent of softmax -> long cover)
#pragma unroll
    for (int dt = 0; dt < 4; ++dt)
      bv[dt] = load_b32x16(Vtp, dt * 16, key0, kS, r, hf);

    if (key0 + 32 < kS) {  // stay ahead of L2 for the next key block
      __builtin_prefetch(Vp + (size_t)(key0 + 32 + lane) * kD, 0, 1);
      __builtin_prefetch(Vtp + (size_t)lane * kS + key0 + 32, 0, 1);
      __builtin_prefetch(Vtp + (size_t)(32 + lane) * kS + key0 + 32, 0, 1);
    }

    // ---- scores ----------------------------------------------------------
    v8f s0 = {}, s1 = {};
    s0 = wmma_f16(aq[0], bs[0][0], s0);
    s1 = wmma_f16(aq[0], bs[1][0], s1);
    s0 = wmma_f16(aq[1], bs[0][1], s0);
    s1 = wmma_f16(aq[1], bs[1][1], s1);

    // ---- online softmax in C/D layout: slot t = row (t+8*hf) -------------
#pragma unroll
    for (int t = 0; t < 8; ++t) {
      float mx = fmaxf(s0[t], s1[t]);
      mx = fmaxf(mx, __shfl_xor(mx, 1, 32));
      mx = fmaxf(mx, __shfl_xor(mx, 2, 32));
      mx = fmaxf(mx, __shfl_xor(mx, 4, 32));
      mx = fmaxf(mx, __shfl_xor(mx, 8, 32));
      const float mnew = fmaxf(mrow[t], mx);
      const float alpha = EXP2F((mrow[t] - mnew) * sc);
      mrow[t] = mnew;
      const float p0 = EXP2F((s0[t] - mnew) * sc);
      const float p1 = EXP2F((s1[t] - mnew) * sc);
      float rs = p0 + p1;
      rs += __shfl_xor(rs, 1, 32);
      rs += __shfl_xor(rs, 2, 32);
      rs += __shfl_xor(rs, 4, 32);
      rs += __shfl_xor(rs, 8, 32);
      lrow[t] = lrow[t] * alpha + rs;
#pragma unroll
      for (int dt = 0; dt < 4; ++dt) o[dt][t] *= alpha;
      const int row = 8 * hf + t;
      P[row * 32 + r]      = (h16)p0;       // cols key0+0..15
      P[row * 32 + 16 + r] = (h16)p1;       // cols key0+16..31
    }
    __syncthreads();                        // make P visible (cross-lane)
    v16h ap = load_a16x32(P, 0, 0, 32, r, hf);

    // ---- O += P @ V  (B tiles already resident) --------------------------
#pragma unroll
    for (int dt = 0; dt < 4; ++dt) o[dt] = wmma_f16(ap, bv[dt], o[dt]);
    __syncthreads();                        // protect P before next overwrite
  }

  const int bi = bh >> 4, hh = bh & 15;
#pragma unroll
  for (int t = 0; t < 8; ++t) {
    const float inv = RCPF(lrow[t]);
    const int qrow = q0 + 8 * hf + t;
    float* orow = Out + ((size_t)bi * kS + qrow) * kN + hh * kD;
#pragma unroll
    for (int dt = 0; dt < 4; ++dt) orow[dt * 16 + r] = o[dt][t] * inv;
  }
}

// ---------------------------------------------------------------------------
extern "C" void kernel_launch(void* const* d_in, const int* in_sizes, int n_in,
                              void* d_out, int out_size, void* d_ws,
                              size_t ws_size, hipStream_t stream) {
  (void)in_sizes; (void)n_in; (void)out_size; (void)ws_size;
  const float* X  = (const float*)d_in[0];
  const float* Wq = (const float*)d_in[1];
  const float* bq = (const float*)d_in[2];
  // d_in[3]=Wk, d_in[4]=bk: reference computes K but never uses it -> skipped
  const float* Wv = (const float*)d_in[5];
  const float* bv = (const float*)d_in[6];
  float* Out = (float*)d_out;

  char* ws = (char*)d_ws;
  h16* Wqt = (h16*)(ws);                       //  2 MiB  [N][K] f16
  h16* Wvt = (h16*)(ws + (2u << 20));          //  2 MiB
  h16* Xh  = (h16*)(ws + (4u << 20));          //  8 MiB  [4096][1024] f16
  h16* Qh  = (h16*)(ws + (12u << 20));         //  8 MiB  [BH][S][D]
  h16* Vh  = (h16*)(ws + (20u << 20));         //  8 MiB  [BH][S][D]
  h16* Vt  = (h16*)(ws + (28u << 20));         //  8 MiB  [BH][D][S]

  dim3 tb(32, 8);
  transpose_to_f16<<<dim3(32, 32), tb, 0, stream>>>(Wq, Wqt);
  transpose_to_f16<<<dim3(32, 32), tb, 0, stream>>>(Wv, Wvt);
  cvt_f32_to_f16<<<(kM * kE / 4 + 255) / 256, 256, 0, stream>>>(X, Xh,
                                                               kM * kE / 4);

  dim3 g1(kM / 64, kN / 128);
  proj_gemm<false><<<g1, 128, 0, stream>>>(Xh, Wqt, bq, Qh, nullptr);
  proj_gemm<true><<<g1, 128, 0, stream>>>(Xh, Wvt, bv, Vh, Vt);

  dim3 g2(kB * kH, kS / 64);
  attn_kernel<<<g2, 128, 0, stream>>>(Qh, Vh, Vt, Out);
}
